// SimpleGNN_53274774340236
// MI455X (gfx1250) — matrix operand order repaired and checked by reading, not compiled, not run on hardware
//
#include <hip/hip_runtime.h>

typedef __bf16 bf16;
typedef __attribute__((ext_vector_type(16))) __bf16 v16bf;
typedef __attribute__((ext_vector_type(8)))  __bf16 v8bf;
typedef __attribute__((ext_vector_type(8)))  float  v8f;

#define B_ 16
#define P_ 2048
#define H_ 128
#define K_ 8

// Use gfx1250 async global->LDS copy instructions for kNN candidate staging.
#define USE_ASYNC_STAGE 1

// ---------------------------------------------------------------------------
// WMMA helpers (CDNA5 wave32, 16x16x32 bf16 -> f32)
// ---------------------------------------------------------------------------
__device__ __forceinline__ v8f wmma_bf16(v16bf a, v16bf b, v8f c) {
  return __builtin_amdgcn_wmma_f32_16x16x32_bf16(
      /*neg_a=*/false, a, /*neg_b=*/false, b,
      /*c_mod=*/(short)0, c, /*reuse_a=*/false, /*reuse_b=*/false);
}

// A-matrix 16x32 tile load from row-major [row][K] (stride in elements).
__device__ __forceinline__ v16bf load_a16x32(const bf16* base, int stride,
                                             int row0, int kk, int lane) {
  int r  = row0 + (lane & 15);
  int kb = kk + ((lane & 16) ? 8 : 0);
  const bf16* p = base + (size_t)r * stride + kb;
  v8bf lo = *(const v8bf*)(p);
  v8bf hi = *(const v8bf*)(p + 16);
  return __builtin_shufflevector(lo, hi, 0,1,2,3,4,5,6,7,8,9,10,11,12,13,14,15);
}

// B-matrix 32x16 tile load from row-major [N][K] (pre-transposed weight/features).
__device__ __forceinline__ v16bf load_b32x16(const bf16* base, int stride,
                                             int n0, int kk, int lane) {
  int r  = n0 + (lane & 15);
  int kb = kk + ((lane & 16) ? 8 : 0);
  const bf16* p = base + (size_t)r * stride + kb;
  v8bf lo = *(const v8bf*)(p);
  v8bf hi = *(const v8bf*)(p + 16);
  return __builtin_shufflevector(lo, hi, 0,1,2,3,4,5,6,7,8,9,10,11,12,13,14,15);
}

// register top-8 (ascending d2) insertion network
__device__ __forceinline__ void topk_insert(float d, int j, float dist[8], int id[8]) {
  if (d >= dist[7]) return;
#pragma unroll
  for (int s = 0; s < 8; ++s) {
    bool c = d < dist[s];
    float td = dist[s]; int ti = id[s];
    dist[s] = c ? d : dist[s];
    id[s]   = c ? j : id[s];
    d = c ? td : d;
    j = c ? ti : j;
  }
}

// ---------------------------------------------------------------------------
// Weight prep: dst[h][e] = src[e][h] as bf16, zero-padded to kpad columns
// ---------------------------------------------------------------------------
__global__ void k_prep_w(const float* __restrict__ src, bf16* __restrict__ dst,
                         int ein, int hcols, int kpad) {
  int h = blockIdx.x;
  for (int e = threadIdx.x; e < kpad; e += blockDim.x) {
    float v = (e < ein) ? src[(size_t)e * hcols + h] : 0.0f;
    dst[(size_t)h * kpad + e] = (bf16)v;
  }
}

// ---------------------------------------------------------------------------
// Squared norms of bf16 feature rows
// ---------------------------------------------------------------------------
__global__ void k_norms(const bf16* __restrict__ xb, float* __restrict__ nrm) {
  int p = blockIdx.x * blockDim.x + threadIdx.x;   // over B_*P_
  const bf16* r = xb + (size_t)p * H_;
  float s = 0.f;
#pragma unroll
  for (int c = 0; c < H_; c += 8) {
    v8bf v = *(const v8bf*)(r + c);
#pragma unroll
    for (int i = 0; i < 8; ++i) { float f = (float)v[i]; s += f * f; }
  }
  nrm[p] = s;
}

// ---------------------------------------------------------------------------
// kNN on raw xyz (dim 3), scalar
// ---------------------------------------------------------------------------
__global__ __launch_bounds__(256) void k_knn3(const float* __restrict__ x,
                                              int* __restrict__ idx) {
  __shared__ float pts[P_ * 3];
  int b = blockIdx.y;
  const float* xb = x + (size_t)b * P_ * 3;
  for (int i = threadIdx.x; i < P_ * 3; i += 256) pts[i] = xb[i];
  __syncthreads();
  int q = blockIdx.x * 256 + threadIdx.x;
  float qx = pts[q * 3 + 0], qy = pts[q * 3 + 1], qz = pts[q * 3 + 2];
  float dist[8]; int id[8];
#pragma unroll
  for (int i = 0; i < 8; ++i) { dist[i] = 1e30f; id[i] = 0; }
  for (int j = 0; j < P_; ++j) {
    float dx = pts[j * 3 + 0] - qx;
    float dy = pts[j * 3 + 1] - qy;
    float dz = pts[j * 3 + 2] - qz;
    float d = dx * dx + dy * dy + dz * dz;
    if (j == q) d = -1e30f;           // self always first (matches top_k(-d2))
    topk_insert(d, j, dist, id);
  }
  int* o = idx + ((size_t)b * P_ + q) * K_;
#pragma unroll
  for (int k = 0; k < 8; ++k) o[k] = id[k];
}

// ---------------------------------------------------------------------------
// kNN on 128-dim bf16 features via WMMA Gram tiles.
// grid: (P_/128, B_), block 256 (8 waves). Candidate tiles of 64 are staged
// into LDS once per workgroup (async global->LDS copy with per-row padding),
// then all 8 waves read B tiles from LDS.
// ---------------------------------------------------------------------------
__global__ __launch_bounds__(256) void k_knn_wmma(const bf16* __restrict__ xb,
                                                  const float* __restrict__ nrm,
                                                  int* __restrict__ idx) {
  __shared__ alignas(16) float S[128 * 68];     // 34816 B, padded stride
  __shared__ alignas(16) bf16  Sc[64 * 136];    // 17408 B, candidate stage, padded
  int b = blockIdx.y;
  int q0 = blockIdx.x * 128;
  const bf16* F = xb + (size_t)b * P_ * H_;
  const float* N = nrm + (size_t)b * P_;
  int tid = threadIdx.x, lane = tid & 31, w = tid >> 5;
  int rb = q0 + w * 16;

  v16bf A[4];
#pragma unroll
  for (int ks = 0; ks < 4; ++ks) A[ks] = load_a16x32(F, H_, rb, ks * 32, lane);

  float dist[8]; int id[8];
#pragma unroll
  for (int i = 0; i < 8; ++i) { dist[i] = 1e30f; id[i] = 0; }
  float nq = N[q0 + (tid & 127)];

  for (int ct = 0; ct < P_ / 64; ++ct) {
    int c0 = ct * 64;
    // ---- stage 64 candidate rows (contiguous 16 KB) into padded LDS ----
#pragma unroll
    for (int i = 0; i < 4; ++i) {
      int chunk = tid + i * 256;          // 1024 chunks of 16 B
      int row = chunk >> 4;               // 16 chunks per 256 B feature row
      int c16 = chunk & 15;
      const bf16* g = F + (size_t)c0 * H_ + (size_t)chunk * 8;
#if USE_ASYNC_STAGE
      unsigned ldsoff = (unsigned)(size_t)Sc + (unsigned)(row * 272 + c16 * 16);
      asm volatile("global_load_async_to_lds_b128 %0, %1, off"
                   :: "v"(ldsoff), "v"(g) : "memory");
#else
      uint4 v = *(const uint4*)g;
      *(uint4*)((char*)Sc + row * 272 + c16 * 16) = v;
#endif
    }
#if USE_ASYNC_STAGE
    asm volatile("s_wait_asynccnt 0x0" ::: "memory");
#endif
    __syncthreads();

    // ---- Gram tiles from LDS-staged candidates ----
#pragma unroll
    for (int n = 0; n < 4; ++n) {
      v16bf Bt[4];
#pragma unroll
      for (int ks = 0; ks < 4; ++ks)
        Bt[ks] = load_b32x16(Sc, 136, n * 16, ks * 32, lane);
      v8f acc = {0.f, 0.f, 0.f, 0.f, 0.f, 0.f, 0.f, 0.f};
#pragma unroll
      for (int ks = 0; ks < 4; ++ks) acc = wmma_bf16(A[ks], Bt[ks], acc);
      int col = n * 16 + (lane & 15);
      int m0  = w * 16 + ((lane & 16) ? 8 : 0);
#pragma unroll
      for (int e = 0; e < 8; ++e) S[(m0 + e) * 68 + col] = acc[e];
    }
    __syncthreads();

    // ---- top-8 scan ----
    if (tid < 128) {
      int q = q0 + tid;
      for (int jj = 0; jj < 64; ++jj) {
        int jc = (jj + tid) & 63;                  // rotate to spread LDS banks
        float s = S[tid * 68 + jc];
        int j = c0 + jc;
        float d = nq + N[j] - 2.0f * s;
        if (j == q) d = -1e30f;
        topk_insert(d, j, dist, id);
      }
    }
    __syncthreads();   // scan done before next stage overwrites Sc/S
  }
  if (tid < 128) {
    int* o = idx + ((size_t)b * P_ + q0 + tid) * K_;
#pragma unroll
    for (int k = 0; k < 8; ++k) o[k] = id[k];
  }
}

// ---------------------------------------------------------------------------
// EdgeConv1: e = [xi(3), xj-xi(3)] padded to K=32, MLP 32->128->128, mean over 8
// grid (P_/16, B_), block 256 (8 waves), 128 edge-rows per workgroup
// ---------------------------------------------------------------------------
__global__ __launch_bounds__(256) void k_edge1(
    const float* __restrict__ x, const int* __restrict__ idx,
    const bf16* __restrict__ W1t, const float* __restrict__ b1,
    const bf16* __restrict__ W2t, const float* __restrict__ b2,
    float* __restrict__ outf, bf16* __restrict__ outb) {
  __shared__ alignas(16) bf16 E[128 * 40];     // 10.0 KB
  __shared__ alignas(16) bf16 H1[128 * 136];   // 34.8 KB
  int b = blockIdx.y, p0 = blockIdx.x * 16;
  int tid = threadIdx.x, lane = tid & 31, w = tid >> 5;
  const float* xbase = x + (size_t)b * P_ * 3;

  {  // gather
    int r = tid & 127, half = tid >> 7;
    int p = p0 + (r >> 3);
    int j = idx[((size_t)b * P_ + p) * K_ + (r & 7)];
    bf16* e = E + r * 40 + half * 16;
    if (half == 0) {
      float xi0 = xbase[p * 3 + 0], xi1 = xbase[p * 3 + 1], xi2 = xbase[p * 3 + 2];
      float xj0 = xbase[j * 3 + 0], xj1 = xbase[j * 3 + 1], xj2 = xbase[j * 3 + 2];
      e[0] = (bf16)xi0; e[1] = (bf16)xi1; e[2] = (bf16)xi2;
      e[3] = (bf16)(xj0 - xi0); e[4] = (bf16)(xj1 - xi1); e[5] = (bf16)(xj2 - xi2);
#pragma unroll
      for (int c = 6; c < 16; ++c) e[c] = (bf16)0.f;
    } else {
#pragma unroll
      for (int c = 0; c < 16; ++c) e[c] = (bf16)0.f;
    }
  }
  __syncthreads();

  int rb = w * 16;
  // GEMM1: (128x32) x (32x128) -- all 8 B tiles preloaded, then WMMA chain
  v16bf a1 = load_a16x32(E, 40, rb, 0, lane);
  {
    v16bf Bt[8];
#pragma unroll
    for (int n = 0; n < 8; ++n) Bt[n] = load_b32x16(W1t, 32, n * 16, 0, lane);
#pragma unroll
    for (int n = 0; n < 8; ++n) {
      v8f acc = {0.f, 0.f, 0.f, 0.f, 0.f, 0.f, 0.f, 0.f};
      acc = wmma_bf16(a1, Bt[n], acc);
      int col = n * 16 + (lane & 15);
      float bias = b1[col];
      int m0 = rb + ((lane & 16) ? 8 : 0);
#pragma unroll
      for (int e = 0; e < 8; ++e) {
        float h = acc[e] + bias; h = h > 0.f ? h : 0.f;
        H1[(m0 + e) * 136 + col] = (bf16)h;
      }
    }
  }
  __syncthreads();

  // GEMM2: (128x128) x (128x128), fused bias+ReLU+mean over 8 neighbors
  v16bf a2[4];
#pragma unroll
  for (int ks = 0; ks < 4; ++ks) a2[ks] = load_a16x32(H1, 136, rb, ks * 32, lane);
#pragma unroll
  for (int n = 0; n < 8; ++n) {
    v16bf Bt[4];
#pragma unroll
    for (int ks = 0; ks < 4; ++ks)
      Bt[ks] = load_b32x16(W2t, 128, n * 16, ks * 32, lane);
    v8f acc = {0.f, 0.f, 0.f, 0.f, 0.f, 0.f, 0.f, 0.f};
#pragma unroll
    for (int ks = 0; ks < 4; ++ks) acc = wmma_bf16(a2[ks], Bt[ks], acc);
    int col = n * 16 + (lane & 15);
    float bias = b2[col];
    float s = 0.f;
#pragma unroll
    for (int e = 0; e < 8; ++e) {
      float h = acc[e] + bias; s += (h > 0.f ? h : 0.f);
    }
    float val = s * 0.125f;   // 8 elements of a lane == 8 neighbors of one point
    int point = p0 + 2 * w + ((lane & 16) ? 1 : 0);
    size_t o = ((size_t)b * P_ + point) * H_ + col;
    outf[o] = val;
    outb[o] = (bf16)val;
  }
}

// ---------------------------------------------------------------------------
// EdgeConv2/3: e = [xi(128), xj-xi(128)], MLP 256->128->128, mean over 8
// grid (P_/8, B_), block 128 (4 waves), 64 edge-rows per workgroup
// ---------------------------------------------------------------------------
__global__ __launch_bounds__(128) void k_edge2(
    const float* __restrict__ xin, const int* __restrict__ idx,
    const bf16* __restrict__ W1t, const float* __restrict__ b1,
    const bf16* __restrict__ W2t, const float* __restrict__ b2,
    float* __restrict__ outf, bf16* __restrict__ outb) {
  __shared__ alignas(16) bf16 E[64 * 264];     // 33.8 KB
  __shared__ alignas(16) bf16 H1[64 * 136];    // 17.4 KB
  int b = blockIdx.y, p0 = blockIdx.x * 8;
  int tid = threadIdx.x, lane = tid & 31, w = tid >> 5;

  {  // gather: 2 threads per edge-row, 128 cols each
    int r = tid >> 1, half = tid & 1;
    int p = p0 + (r >> 3);
    int j = idx[((size_t)b * P_ + p) * K_ + (r & 7)];
    const float* xi = xin + ((size_t)b * P_ + p) * H_;
    const float* xj = xin + ((size_t)b * P_ + j) * H_;
    bf16* e = E + r * 264 + half * 128;
    if (half == 0) {
#pragma unroll
      for (int c = 0; c < 128; c += 4) {
        float4 v = *(const float4*)(xi + c);
        e[c + 0] = (bf16)v.x; e[c + 1] = (bf16)v.y;
        e[c + 2] = (bf16)v.z; e[c + 3] = (bf16)v.w;
      }
    } else {
#pragma unroll
      for (int c = 0; c < 128; c += 4) {
        float4 a = *(const float4*)(xi + c);
        float4 v = *(const float4*)(xj + c);
        e[c + 0] = (bf16)(v.x - a.x); e[c + 1] = (bf16)(v.y - a.y);
        e[c + 2] = (bf16)(v.z - a.z); e[c + 3] = (bf16)(v.w - a.w);
      }
    }
  }
  __syncthreads();

  int rb = w * 16;
  // GEMM1: (64x256) x (256x128)
  v16bf a1[8];
#pragma unroll
  for (int ks = 0; ks < 8; ++ks) a1[ks] = load_a16x32(E, 264, rb, ks * 32, lane);
#pragma unroll
  for (int n = 0; n < 8; ++n) {
    v16bf Bt[8];
#pragma unroll
    for (int ks = 0; ks < 8; ++ks)
      Bt[ks] = load_b32x16(W1t, 256, n * 16, ks * 32, lane);
    v8f acc = {0.f, 0.f, 0.f, 0.f, 0.f, 0.f, 0.f, 0.f};
#pragma unroll
    for (int ks = 0; ks < 8; ++ks) acc = wmma_bf16(a1[ks], Bt[ks], acc);
    int col = n * 16 + (lane & 15);
    float bias = b1[col];
    int m0 = rb + ((lane & 16) ? 8 : 0);
#pragma unroll
    for (int e = 0; e < 8; ++e) {
      float h = acc[e] + bias; h = h > 0.f ? h : 0.f;
      H1[(m0 + e) * 136 + col] = (bf16)h;
    }
  }
  __syncthreads();

  // GEMM2: (64x128) x (128x128), fused bias+ReLU+mean
  v16bf a2[4];
#pragma unroll
  for (int ks = 0; ks < 4; ++ks) a2[ks] = load_a16x32(H1, 136, rb, ks * 32, lane);
#pragma unroll
  for (int n = 0; n < 8; ++n) {
    v16bf Bt[4];
#pragma unroll
    for (int ks = 0; ks < 4; ++ks)
      Bt[ks] = load_b32x16(W2t, 128, n * 16, ks * 32, lane);
    v8f acc = {0.f, 0.f, 0.f, 0.f, 0.f, 0.f, 0.f, 0.f};
#pragma unroll
    for (int ks = 0; ks < 4; ++ks) acc = wmma_bf16(a2[ks], Bt[ks], acc);
    int col = n * 16 + (lane & 15);
    float bias = b2[col];
    float s = 0.f;
#pragma unroll
    for (int e = 0; e < 8; ++e) {
      float h = acc[e] + bias; s += (h > 0.f ? h : 0.f);
    }
    float val = s * 0.125f;
    int point = p0 + 2 * w + ((lane & 16) ? 1 : 0);
    size_t o = ((size_t)b * P_ + point) * H_ + col;
    outf[o] = val;
    outb[o] = (bf16)val;
  }
}

// ---------------------------------------------------------------------------
// Global mean over points of concat(x1,x2,x3): grid B_, block 384
// ---------------------------------------------------------------------------
__global__ void k_pool(const float* __restrict__ x1, const float* __restrict__ x2,
                       const float* __restrict__ x3, float* __restrict__ pooled) {
  int b = blockIdx.x, f = threadIdx.x;
  const float* src = (f < 128) ? x1 : (f < 256 ? x2 : x3);
  int c = f & 127;
  const float* p = src + (size_t)b * P_ * H_ + c;
  float s = 0.f;
  for (int i = 0; i < P_; ++i) s += p[(size_t)i * H_];
  pooled[b * 384 + f] = s * (1.0f / P_);
}

// ---------------------------------------------------------------------------
// Head MLP: relu(pooled @ mW1 + mb1) @ mW2 + mb2 -> (16,2)
// ---------------------------------------------------------------------------
__global__ void k_mlp(const float* __restrict__ pooled,
                      const float* __restrict__ W1, const float* __restrict__ bb1,
                      const float* __restrict__ W2, const float* __restrict__ bb2,
                      float* __restrict__ out) {
  __shared__ float h[16 * 128];
  int tid = threadIdx.x;
  for (int t = tid; t < 16 * 128; t += 256) {
    int b = t >> 7, hh = t & 127;
    float s = bb1[hh];
    for (int e = 0; e < 384; ++e) s += pooled[b * 384 + e] * W1[e * 128 + hh];
    h[t] = s > 0.f ? s : 0.f;
  }
  __syncthreads();
  if (tid < 32) {
    int b = tid >> 1, o = tid & 1;
    float s = bb2[o];
    for (int e = 0; e < 128; ++e) s += h[b * 128 + e] * W2[e * 2 + o];
    out[b * 2 + o] = s;
  }
}

// ---------------------------------------------------------------------------
extern "C" void kernel_launch(void* const* d_in, const int* in_sizes, int n_in,
                              void* d_out, int out_size, void* d_ws, size_t ws_size,
                              hipStream_t stream) {
  (void)in_sizes; (void)n_in; (void)out_size; (void)ws_size;
  const float* x    = (const float*)d_in[0];
  const float* c1W1 = (const float*)d_in[1];
  const float* c1b1 = (const float*)d_in[2];
  const float* c1W2 = (const float*)d_in[3];
  const float* c1b2 = (const float*)d_in[4];
  const float* c2W1 = (const float*)d_in[5];
  const float* c2b1 = (const float*)d_in[6];
  const float* c2W2 = (const float*)d_in[7];
  const float* c2b2 = (const float*)d_in[8];
  const float* c3W1 = (const float*)d_in[9];
  const float* c3b1 = (const float*)d_in[10];
  const float* c3W2 = (const float*)d_in[11];
  const float* c3b2 = (const float*)d_in[12];
  const float* mW1  = (const float*)d_in[13];
  const float* mb1  = (const float*)d_in[14];
  const float* mW2  = (const float*)d_in[15];
  const float* mb2  = (const float*)d_in[16];

  char* wsp = (char*)d_ws;
  size_t off = 0;
  auto take = [&](size_t n) -> char* {
    char* p = wsp + off;
    off = (off + n + 255) & ~(size_t)255;
    return p;
  };
  const size_t featF = (size_t)B_ * P_ * H_ * sizeof(float);
  const size_t featB = (size_t)B_ * P_ * H_ * sizeof(bf16);
  float* x1f = (float*)take(featF);
  float* x2f = (float*)take(featF);
  float* x3f = (float*)take(featF);
  bf16*  x1b = (bf16*)take(featB);
  bf16*  x2b = (bf16*)take(featB);
  bf16*  x3b = (bf16*)take(featB);
  float* n1  = (float*)take((size_t)B_ * P_ * sizeof(float));
  float* n2  = (float*)take((size_t)B_ * P_ * sizeof(float));
  int* idx1  = (int*)take((size_t)B_ * P_ * K_ * sizeof(int));
  int* idx2  = (int*)take((size_t)B_ * P_ * K_ * sizeof(int));
  int* idx3  = (int*)take((size_t)B_ * P_ * K_ * sizeof(int));
  bf16* w1t1 = (bf16*)take(128 * 32 * sizeof(bf16));
  bf16* w2t1 = (bf16*)take(128 * 128 * sizeof(bf16));
  bf16* w1t2 = (bf16*)take(128 * 256 * sizeof(bf16));
  bf16* w2t2 = (bf16*)take(128 * 128 * sizeof(bf16));
  bf16* w1t3 = (bf16*)take(128 * 256 * sizeof(bf16));
  bf16* w2t3 = (bf16*)take(128 * 128 * sizeof(bf16));
  float* pooled = (float*)take(16 * 384 * sizeof(float));

  // weight prep (transpose -> bf16, pad conv1 edge dim 6 -> 32)
  k_prep_w<<<128, 64, 0, stream>>>(c1W1, w1t1, 6,   128, 32);
  k_prep_w<<<128, 64, 0, stream>>>(c1W2, w2t1, 128, 128, 128);
  k_prep_w<<<128, 64, 0, stream>>>(c2W1, w1t2, 256, 128, 256);
  k_prep_w<<<128, 64, 0, stream>>>(c2W2, w2t2, 128, 128, 128);
  k_prep_w<<<128, 64, 0, stream>>>(c3W1, w1t3, 256, 128, 256);
  k_prep_w<<<128, 64, 0, stream>>>(c3W2, w2t3, 128, 128, 128);

  // layer 1
  k_knn3<<<dim3(P_ / 256, B_), 256, 0, stream>>>(x, idx1);
  k_edge1<<<dim3(P_ / 16, B_), 256, 0, stream>>>(x, idx1, w1t1, c1b1, w2t1, c1b2, x1f, x1b);

  // layer 2
  k_norms<<<(B_ * P_) / 256, 256, 0, stream>>>(x1b, n1);
  k_knn_wmma<<<dim3(P_ / 128, B_), 256, 0, stream>>>(x1b, n1, idx2);
  k_edge2<<<dim3(P_ / 8, B_), 128, 0, stream>>>(x1f, idx2, w1t2, c2b1, w2t2, c2b2, x2f, x2b);

  // layer 3
  k_norms<<<(B_ * P_) / 256, 256, 0, stream>>>(x2b, n2);
  k_knn_wmma<<<dim3(P_ / 128, B_), 256, 0, stream>>>(x2b, n2, idx3);
  k_edge2<<<dim3(P_ / 8, B_), 128, 0, stream>>>(x2f, idx3, w1t3, c3b1, w2t3, c3b2, x3f, x3b);

  // head
  k_pool<<<B_, 384, 0, stream>>>(x1f, x2f, x3f, pooled);
  k_mlp<<<1, 256, 0, stream>>>(pooled, mW1, mb1, mW2, mb2, (float*)d_out);
}
